// FallbackMessagePassingLayer_10496900071648
// MI455X (gfx1250) — matrix operand order repaired and checked by reading, not compiled, not run on hardware
//
#include <hip/hip_runtime.h>

typedef __attribute__((ext_vector_type(16))) _Float16 v16h;
typedef __attribute__((ext_vector_type(8)))  _Float16 v8h;
typedef __attribute__((ext_vector_type(8)))  float    v8f;
typedef __attribute__((ext_vector_type(4)))  float    v4f;

static constexpr int kN     = 50000;   // nodes
static constexpr int kE     = 800000;  // edges
static constexpr int kH     = 64;      // hidden
static constexpr int kED    = 32;      // edge feature dim
static constexpr int kWAVES = 8;       // waves per block (wave32)
static constexpr int kBLK   = kWAVES * 32;

#define WMMA16(a, b, c) \
  __builtin_amdgcn_wmma_f32_16x16x32_f16(false, (a), false, (b), (short)0, (c), false, false)

__device__ __forceinline__ float silu_f(float v) { return v / (1.0f + __expf(-v)); }

// ---------------------------------------------------------------------------
// zero the aggregation buffer (d_out reused as agg accumulator)
// ---------------------------------------------------------------------------
__global__ void zero_f32(float* __restrict__ p, int n4) {
  v4f z = {0.f, 0.f, 0.f, 0.f};
  for (int i = blockIdx.x * blockDim.x + threadIdx.x; i < n4;
       i += gridDim.x * blockDim.x)
    ((v4f*)p)[i] = z;
}

// ---------------------------------------------------------------------------
// Pack helper: store weight [Krows x 64] (f32, row-major) into LDS as f16
// B-fragments. Fragment (s,t): lane L holds column N=16t+(L&15),
// elements e=0..15 are K = 32s + (e>>3)*16 + (L>>4)*8 + (e&7).  512 f16 per fragment,
// lane chunk contiguous at frag*512 + L*16.
// ---------------------------------------------------------------------------
__device__ __forceinline__ void pack_bfrags(_Float16* dst, const float* w,
                                            int nfrag, int tid) {
  for (int idx = tid; idx < nfrag * 512; idx += kBLK) {
    int frag = idx >> 9;
    int s    = frag >> 2;       // k-step
    int t    = frag & 3;        // n-tile
    int ln   = (idx >> 4) & 31;
    int e    = idx & 15;
    int K    = 32 * s + ((e >> 3) << 4) + ((ln >> 4) << 3) + (e & 7);
    int Nc   = 16 * t + (ln & 15);
    dst[idx] = (_Float16)w[K * kH + Nc];
  }
}

// ---------------------------------------------------------------------------
// Message MLP over edges + scatter-add into agg:
//   m = silu([x[src]|x[tgt]|ea] @ w1 + b1) @ w2 + b2 ;  agg[tgt] += m
// One wave handles a 16-edge tile.
// ---------------------------------------------------------------------------
__global__ __launch_bounds__(kBLK) void msg_kernel(
    const float* __restrict__ x, const long long* __restrict__ eidx,
    const float* __restrict__ ea, const float* __restrict__ w1,
    const float* __restrict__ b1, const float* __restrict__ w2,
    const float* __restrict__ b2, float* __restrict__ agg) {
  __shared__ __align__(32) _Float16 w1f[20 * 512];        // 20 KB
  __shared__ __align__(32) _Float16 w2f[8 * 512];         //  8 KB
  __shared__ __align__(32) _Float16 stage[kWAVES][1024];  // 16 KB

  const int tid = threadIdx.x;
  pack_bfrags(w1f, w1, 20, tid);  // K=160 -> 5 k-steps x 4 n-tiles
  pack_bfrags(w2f, w2, 8, tid);   // K=64  -> 2 k-steps x 4 n-tiles
  __syncthreads();

  const int lane = tid & 31;
  const int wv   = tid >> 5;
  const int hf   = lane >> 4;   // which 16-lane half
  const int r16  = lane & 15;
  const int wgid    = blockIdx.x * kWAVES + wv;
  const int wstride = gridDim.x * kWAVES;

  float b1v[4], b2v[4];
#pragma unroll
  for (int t = 0; t < 4; ++t) {
    b1v[t] = b1[r16 + 16 * t];
    b2v[t] = b2[r16 + 16 * t];
  }
  _Float16* st = stage[wv];

  for (int tile = wgid; tile < kE / 16; tile += wstride) {
    const int e0  = tile * 16;
    const int row = e0 + r16;            // this lane's A-matrix row (edge id)
    const int src = (int)eidx[row];
    const int tg  = (int)eidx[kE + row];

    // ----- layer 1: [16x160] @ [160x64], bias in accumulator -----
    v8f acc[4];
#pragma unroll
    for (int t = 0; t < 4; ++t)
#pragma unroll
      for (int r = 0; r < 8; ++r) acc[t][r] = b1v[t];

#pragma unroll
    for (int s = 0; s < 5; ++s) {
      const float* base;
      if (s < 2)       base = x + (size_t)src * kH + 32 * s;
      else if (s < 4)  base = x + (size_t)tg * kH + 32 * (s - 2);
      else             base = ea + (size_t)row * kED;
      const float* p0 = base + hf * 8;   // K-runs: kbase..+7 and 16+kbase..+7
      v4f f0 = *(const v4f*)(p0);
      v4f f1 = *(const v4f*)(p0 + 4);
      v4f f2 = *(const v4f*)(p0 + 16);
      v4f f3 = *(const v4f*)(p0 + 20);
      v16h a;
#pragma unroll
      for (int i = 0; i < 4; ++i) {
        a[i]      = (_Float16)f0[i];
        a[4 + i]  = (_Float16)f1[i];
        a[8 + i]  = (_Float16)f2[i];
        a[12 + i] = (_Float16)f3[i];
      }
#pragma unroll
      for (int t = 0; t < 4; ++t) {
        v16h b = *(const v16h*)&w1f[((s * 4 + t) << 9) + lane * 16];
        acc[t] = WMMA16(a, b, acc[t]);
      }
    }

    // ----- SiLU, stage as f16 row-major [16x64] for layer-2 A transpose -----
#pragma unroll
    for (int t = 0; t < 4; ++t)
#pragma unroll
      for (int r = 0; r < 8; ++r)
        st[(r + 8 * hf) * kH + r16 + 16 * t] = (_Float16)silu_f(acc[t][r]);
    asm volatile("" ::: "memory");  // same-wave LDS ops are in-order on HW

    // ----- layer 2: [16x64] @ [64x64] -----
    v8f acc2[4];
#pragma unroll
    for (int t = 0; t < 4; ++t)
#pragma unroll
      for (int r = 0; r < 8; ++r) acc2[t][r] = b2v[t];

#pragma unroll
    for (int s = 0; s < 2; ++s) {
      const _Float16* sp = st + r16 * kH + 32 * s + hf * 8;
      v8h lo = *(const v8h*)sp;
      v8h hi = *(const v8h*)(sp + 16);
      v16h a;
#pragma unroll
      for (int i = 0; i < 8; ++i) {
        a[i]     = lo[i];
        a[8 + i] = hi[i];
      }
#pragma unroll
      for (int t = 0; t < 4; ++t) {
        v16h b  = *(const v16h*)&w2f[((s * 4 + t) << 9) + lane * 16];
        acc2[t] = WMMA16(a, b, acc2[t]);
      }
    }

    // ----- scatter-add m rows into agg[tgt] -----
    int tgm[8];
#pragma unroll
    for (int r = 0; r < 8; ++r) tgm[r] = __shfl(tg, r + 8 * hf, 32);
#pragma unroll
    for (int t = 0; t < 4; ++t)
#pragma unroll
      for (int r = 0; r < 8; ++r)
        atomicAdd(&agg[(size_t)tgm[r] * kH + r16 + 16 * t], acc2[t][r]);
  }
}

// ---------------------------------------------------------------------------
// Update MLP + residual + LayerNorm:
//   out = LN(x + silu([x|agg] @ w3 + b3) @ w4 + b4)
// `out` holds agg on entry; each 16-node tile reads its agg rows before
// overwriting them (row-local, tile-exclusive -> safe).
// ---------------------------------------------------------------------------
__global__ __launch_bounds__(kBLK) void upd_kernel(
    const float* __restrict__ x, const float* __restrict__ w3,
    const float* __restrict__ b3, const float* __restrict__ w4,
    const float* __restrict__ b4, const float* __restrict__ gamma,
    const float* __restrict__ beta, float* __restrict__ out) {
  __shared__ __align__(32) _Float16 w3f[16 * 512];        // 16 KB
  __shared__ __align__(32) _Float16 w4f[8 * 512];         //  8 KB
  __shared__ __align__(32) _Float16 stage[kWAVES][1024];  // 16 KB

  const int tid = threadIdx.x;
  pack_bfrags(w3f, w3, 16, tid);  // K=128 -> 4 k-steps x 4 n-tiles
  pack_bfrags(w4f, w4, 8, tid);
  __syncthreads();

  const int lane = tid & 31;
  const int wv   = tid >> 5;
  const int hf   = lane >> 4;
  const int r16  = lane & 15;
  const int wgid    = blockIdx.x * kWAVES + wv;
  const int wstride = gridDim.x * kWAVES;

  float b3v[4], b4v[4], gv[4], bv[4];
#pragma unroll
  for (int t = 0; t < 4; ++t) {
    b3v[t] = b3[r16 + 16 * t];
    b4v[t] = b4[r16 + 16 * t];
    gv[t]  = gamma[r16 + 16 * t];
    bv[t]  = beta[r16 + 16 * t];
  }
  _Float16* st = stage[wv];

  for (int tile = wgid; tile < kN / 16; tile += wstride) {
    const int n0  = tile * 16;
    const int row = n0 + r16;

    v8f acc[4];
#pragma unroll
    for (int t = 0; t < 4; ++t)
#pragma unroll
      for (int r = 0; r < 8; ++r) acc[t][r] = b3v[t];

#pragma unroll
    for (int s = 0; s < 4; ++s) {
      const float* base = (s < 2) ? x + (size_t)row * kH + 32 * s
                                  : out + (size_t)row * kH + 32 * (s - 2);
      const float* p0 = base + hf * 8;
      v4f f0 = *(const v4f*)(p0);
      v4f f1 = *(const v4f*)(p0 + 4);
      v4f f2 = *(const v4f*)(p0 + 16);
      v4f f3 = *(const v4f*)(p0 + 20);
      v16h a;
#pragma unroll
      for (int i = 0; i < 4; ++i) {
        a[i]      = (_Float16)f0[i];
        a[4 + i]  = (_Float16)f1[i];
        a[8 + i]  = (_Float16)f2[i];
        a[12 + i] = (_Float16)f3[i];
      }
#pragma unroll
      for (int t = 0; t < 4; ++t) {
        v16h b = *(const v16h*)&w3f[((s * 4 + t) << 9) + lane * 16];
        acc[t] = WMMA16(a, b, acc[t]);
      }
    }

#pragma unroll
    for (int t = 0; t < 4; ++t)
#pragma unroll
      for (int r = 0; r < 8; ++r)
        st[(r + 8 * hf) * kH + r16 + 16 * t] = (_Float16)silu_f(acc[t][r]);
    asm volatile("" ::: "memory");

    v8f acc2[4];
#pragma unroll
    for (int t = 0; t < 4; ++t)
#pragma unroll
      for (int r = 0; r < 8; ++r) acc2[t][r] = b4v[t];

#pragma unroll
    for (int s = 0; s < 2; ++s) {
      const _Float16* sp = st + r16 * kH + 32 * s + hf * 8;
      v8h lo = *(const v8h*)sp;
      v8h hi = *(const v8h*)(sp + 16);
      v16h a;
#pragma unroll
      for (int i = 0; i < 8; ++i) {
        a[i]     = lo[i];
        a[8 + i] = hi[i];
      }
#pragma unroll
      for (int t = 0; t < 4; ++t) {
        v16h b  = *(const v16h*)&w4f[((s * 4 + t) << 9) + lane * 16];
        acc2[t] = WMMA16(a, b, acc2[t]);
      }
    }

    // residual + LayerNorm (row M = r + 8*hf lives in this lane's 16-group)
    float dv[4][8];
#pragma unroll
    for (int t = 0; t < 4; ++t)
#pragma unroll
      for (int r = 0; r < 8; ++r)
        dv[t][r] = x[(size_t)(n0 + r + 8 * hf) * kH + r16 + 16 * t] + acc2[t][r];

#pragma unroll
    for (int r = 0; r < 8; ++r) {
      float sm = dv[0][r] + dv[1][r] + dv[2][r] + dv[3][r];
      float sq = dv[0][r] * dv[0][r] + dv[1][r] * dv[1][r] +
                 dv[2][r] * dv[2][r] + dv[3][r] * dv[3][r];
#pragma unroll
      for (int m = 1; m < 16; m <<= 1) {  // reduce within 16-lane half
        sm += __shfl_xor(sm, m, 32);
        sq += __shfl_xor(sq, m, 32);
      }
      float mean = sm * (1.0f / 64.0f);
      float var  = sq * (1.0f / 64.0f) - mean * mean;
      float rstd = rsqrtf(var + 1e-5f);
#pragma unroll
      for (int t = 0; t < 4; ++t)
        out[(size_t)(n0 + r + 8 * hf) * kH + r16 + 16 * t] =
            (dv[t][r] - mean) * rstd * gv[t] + bv[t];
    }
  }
}

// ---------------------------------------------------------------------------
extern "C" void kernel_launch(void* const* d_in, const int* in_sizes, int n_in,
                              void* d_out, int out_size, void* d_ws,
                              size_t ws_size, hipStream_t stream) {
  (void)in_sizes; (void)n_in; (void)out_size; (void)d_ws; (void)ws_size;
  const float*     x     = (const float*)d_in[0];
  const long long* eidx  = (const long long*)d_in[1];  // int64 [2,E]
  const float*     ea    = (const float*)d_in[2];
  const float*     w1    = (const float*)d_in[3];
  const float*     b1    = (const float*)d_in[4];
  const float*     w2    = (const float*)d_in[5];
  const float*     b2    = (const float*)d_in[6];
  const float*     w3    = (const float*)d_in[7];
  const float*     b3    = (const float*)d_in[8];
  const float*     w4    = (const float*)d_in[9];
  const float*     b4    = (const float*)d_in[10];
  const float*     gamma = (const float*)d_in[11];
  const float*     beta  = (const float*)d_in[12];
  float*           out   = (float*)d_out;

  // 1) zero agg (d_out doubles as agg accumulator)
  zero_f32<<<1024, 256, 0, stream>>>(out, kN * kH / 4);
  // 2) edge message MLP + scatter-add (50000 edge tiles)
  msg_kernel<<<1024, kBLK, 0, stream>>>(x, eidx, ea, w1, b1, w2, b2, out);
  // 3) node update MLP + residual + LayerNorm (3125 node tiles)
  upd_kernel<<<400, kBLK, 0, stream>>>(x, w3, b3, w4, b4, gamma, beta, out);
}
